// LoRASelfAttention_26139170963759
// MI455X (gfx1250) — matrix operand order repaired and checked by reading, not compiled, not run on hardware
//
#include <hip/hip_runtime.h>
#include <math.h>

// ---------------------------------------------------------------------------
// Types for CDNA5 WMMA (wave32): v16bf A/B fragments, v8f C/D accumulators
// ---------------------------------------------------------------------------
typedef __attribute__((ext_vector_type(16))) __bf16 v16bf;
typedef __attribute__((ext_vector_type(8)))  __bf16 v8bf;
typedef __attribute__((ext_vector_type(8)))  float  v8f;

union Frag {
    v16bf v;
    v8bf  h[2];
};

#define EMBED 1024
#define MTOT  4096        // N*S = 2*2048
#define SEQ   2048
#define HEADS 16
#define HDIM  64
#define L2E   1.44269504088896340736f

// ---------------------------------------------------------------------------
// Kernel 1: fp32 -> bf16 cast of x (row-major [4096][1024])
// ---------------------------------------------------------------------------
__global__ __launch_bounds__(256) void cvt_bf16_kernel(const float* __restrict__ src,
                                                       __bf16* __restrict__ dst, int n)
{
    int i = (blockIdx.x * 256 + threadIdx.x) * 4;
    if (i + 3 < n) {
        float4 f = *(const float4*)(src + i);
        dst[i + 0] = (__bf16)f.x;
        dst[i + 1] = (__bf16)f.y;
        dst[i + 2] = (__bf16)f.z;
        dst[i + 3] = (__bf16)f.w;
    }
}

// ---------------------------------------------------------------------------
// Kernel 2: fold rank-8 LoRA into effective weight:
//   Weff[n][k] = W[n][k] + 2 * sum_r B[n][r] * A[r][k]    (bf16 output)
// ---------------------------------------------------------------------------
__global__ __launch_bounds__(256) void make_weff_kernel(const float* __restrict__ W,
                                                        const float* __restrict__ A,
                                                        const float* __restrict__ B,
                                                        __bf16* __restrict__ Weff)
{
    int idx = blockIdx.x * 256 + threadIdx.x;       // 1M threads, one per (n,k)
    int n = idx >> 10;
    int k = idx & 1023;
    float acc = W[idx];
#pragma unroll
    for (int r = 0; r < 8; ++r)
        acc += 2.0f * B[n * 8 + r] * A[r * 1024 + k];
    Weff[idx] = (__bf16)acc;
}

// ---------------------------------------------------------------------------
// Kernel 3: bf16 WMMA GEMM  Y[m][n] = sum_k Xb[m][k] * Wb[n][k] + bias[n]
//   Block = 128 threads (4 waves), block tile 128x128, wave tile 64x64.
//   A/B tiles staged in LDS via double-buffered GLOBAL_LOAD_ASYNC_TO_LDS_B128
//   (ASYNCcnt-tracked), fragments fed with ds_load_b128.
//   mode 0: write f32 row-major (final output)
//   mode 1: write bf16 headed   [b][h][s][d]   (Q, K)
//   mode 2: write bf16 headed-T [b][h][d][s]   (V, pre-transposed for attn@V)
// ---------------------------------------------------------------------------
__global__ __launch_bounds__(128) void gemm_bf16_wmma(const __bf16* __restrict__ Xb,
                                                      const __bf16* __restrict__ Wb,
                                                      const float*  __restrict__ bias,
                                                      float*  __restrict__ outF,
                                                      __bf16* __restrict__ outH,
                                                      int mode)
{
    __shared__ __attribute__((aligned(128))) __bf16 As[2][128][32];  // 16 KB
    __shared__ __attribute__((aligned(128))) __bf16 Bs[2][128][32];  // 16 KB

    const int tid  = threadIdx.x;
    const int lane = tid & 31;
    const int wave = tid >> 5;
    const int rowBlk = blockIdx.y * 128;
    const int colBlk = blockIdx.x * 128;
    const int wrow = (wave >> 1) * 64;   // wave tile origin inside block tile
    const int wcol = (wave & 1) * 64;
    const int lid  = lane & 15;          // M (A frag) / N (B frag)
    const int hi   = lane >> 4;          // lane-half selector

    // Thread t copies one contiguous 64B row of each 128x32 tile (coalesced),
    // async into LDS; instruction offset applies to both LDS and global addr.
    auto issueTile = [&](int buf, int k0) {
        unsigned la = (unsigned)(uintptr_t)&As[buf][tid][0];
        unsigned lb = (unsigned)(uintptr_t)&Bs[buf][tid][0];
        unsigned long long ga =
            (unsigned long long)(uintptr_t)(Xb + (size_t)(rowBlk + tid) * EMBED + k0);
        unsigned long long gb =
            (unsigned long long)(uintptr_t)(Wb + (size_t)(colBlk + tid) * EMBED + k0);
        asm volatile(
            "global_load_async_to_lds_b128 %0, %2, off\n\t"
            "global_load_async_to_lds_b128 %0, %2, off offset:16\n\t"
            "global_load_async_to_lds_b128 %0, %2, off offset:32\n\t"
            "global_load_async_to_lds_b128 %0, %2, off offset:48\n\t"
            "global_load_async_to_lds_b128 %1, %3, off\n\t"
            "global_load_async_to_lds_b128 %1, %3, off offset:16\n\t"
            "global_load_async_to_lds_b128 %1, %3, off offset:32\n\t"
            "global_load_async_to_lds_b128 %1, %3, off offset:48"
            :: "v"(la), "v"(lb), "v"(ga), "v"(gb) : "memory");
    };

    v8f acc[4][4];
#pragma unroll
    for (int mt = 0; mt < 4; ++mt)
#pragma unroll
        for (int nt = 0; nt < 4; ++nt)
            acc[mt][nt] = (v8f){};

    issueTile(0, 0);                                     // prime the pipeline

    const int NSTEP = EMBED / 32;
    for (int it = 0; it < NSTEP; ++it) {
        const int  cur  = it & 1;
        const bool more = (it + 1 < NSTEP);
        if (more) {
            issueTile(cur ^ 1, (it + 1) * 32);
            // in-order completion: <=8 outstanding means current buffer landed
            asm volatile("s_wait_asynccnt 0x8" ::: "memory");
        } else {
            asm volatile("s_wait_asynccnt 0x0" ::: "memory");
        }
        __syncthreads();   // all waves' async copies visible before reading

        Frag a[4], b[4];
#pragma unroll
        for (int t = 0; t < 4; ++t) {
            a[t].h[0] = *(const v8bf*)&As[cur][wrow + t * 16 + lid][hi * 8];
            a[t].h[1] = *(const v8bf*)&As[cur][wrow + t * 16 + lid][16 + hi * 8];
            b[t].h[0] = *(const v8bf*)&Bs[cur][wcol + t * 16 + lid][hi * 8];
            b[t].h[1] = *(const v8bf*)&Bs[cur][wcol + t * 16 + lid][16 + hi * 8];
        }
#pragma unroll
        for (int mt = 0; mt < 4; ++mt)
#pragma unroll
            for (int nt = 0; nt < 4; ++nt)
                acc[mt][nt] = __builtin_amdgcn_wmma_f32_16x16x32_bf16(
                    false, a[mt].v, false, b[nt].v, (short)0, acc[mt][nt], false, false);

        __syncthreads();   // finish reads before buffer is overwritten
    }

    // Epilogue: C/D layout -> N = lane&15, M = r + 8*(lane>>4)
#pragma unroll
    for (int mt = 0; mt < 4; ++mt)
#pragma unroll
        for (int nt = 0; nt < 4; ++nt)
#pragma unroll
            for (int r = 0; r < 8; ++r) {
                int row = rowBlk + wrow + mt * 16 + r + 8 * hi;
                int col = colBlk + wcol + nt * 16 + lid;
                float v = acc[mt][nt][r] + bias[col];
                if (mode == 0) {
                    outF[(size_t)row * EMBED + col] = v;
                } else {
                    int b_ = row >> 11, s_ = row & 2047;
                    int h_ = col >> 6,  d_ = col & 63;
                    size_t idx = (mode == 1)
                        ? ((((size_t)b_ * HEADS + h_) * SEQ + s_) * HDIM + d_)
                        : ((((size_t)b_ * HEADS + h_) * HDIM + d_) * SEQ + s_);
                    outH[idx] = (__bf16)v;
                }
            }
}

// ---------------------------------------------------------------------------
// Kernel 4: flash attention, one wave per (batch,head, 32-query block).
//   Qh,Kh: [b][h][s][64] bf16;  Vt: [b][h][d][s] bf16 (pre-transposed)
//   att out: [b][s][h*64+d] bf16  (row-major 4096x1024, feeds O-projection)
// ---------------------------------------------------------------------------
__global__ __launch_bounds__(32) void flash_attn_kernel(const __bf16* __restrict__ Qh,
                                                        const __bf16* __restrict__ Kh,
                                                        const __bf16* __restrict__ Vt,
                                                        __bf16* __restrict__ att)
{
    __shared__ __attribute__((aligned(64))) __bf16 Pl[32 * 32];

    const int lane = threadIdx.x & 31;
    const int lid  = lane & 15;
    const int hi   = lane >> 4;
    const int nh   = blockIdx.y;
    const int batch = nh >> 4;
    const int head  = nh & 15;
    const int q0    = blockIdx.x * 32;

    const __bf16* Qp = Qh + (size_t)nh * SEQ * HDIM;
    const __bf16* Kp = Kh + (size_t)nh * SEQ * HDIM;
    const __bf16* Vp = Vt + (size_t)nh * HDIM * SEQ;

    // Q fragments: 2 M-tiles x 2 K(d)-steps, loaded once
    Frag qa[2][2];
#pragma unroll
    for (int mt = 0; mt < 2; ++mt)
#pragma unroll
        for (int dk = 0; dk < 2; ++dk) {
            const __bf16* p = Qp + (size_t)(q0 + mt * 16 + lid) * HDIM + dk * 32 + hi * 8;
            qa[mt][dk].h[0] = *(const v8bf*)p;
            qa[mt][dk].h[1] = *(const v8bf*)(p + 16);
        }

    v8f o[2][4];
    float mrun[2][8], lrun[2][8];
#pragma unroll
    for (int mt = 0; mt < 2; ++mt) {
#pragma unroll
        for (int dt = 0; dt < 4; ++dt) o[mt][dt] = (v8f){};
#pragma unroll
        for (int r = 0; r < 8; ++r) { mrun[mt][r] = -1e30f; lrun[mt][r] = 0.0f; }
    }

    for (int kk0 = 0; kk0 < SEQ; kk0 += 32) {
        // ---- S = (Q K^T) * 0.125 : tiles [2 mt][2 kt] ----
        v8f s[2][2];
#pragma unroll
        for (int mt = 0; mt < 2; ++mt)
#pragma unroll
            for (int kt = 0; kt < 2; ++kt) s[mt][kt] = (v8f){};

#pragma unroll
        for (int kt = 0; kt < 2; ++kt)
#pragma unroll
            for (int dk = 0; dk < 2; ++dk) {
                Frag kb;
                const __bf16* kp = Kp + (size_t)(kk0 + kt * 16 + lid) * HDIM + dk * 32 + hi * 8;
                kb.h[0] = *(const v8bf*)kp;
                kb.h[1] = *(const v8bf*)(kp + 16);
#pragma unroll
                for (int mt = 0; mt < 2; ++mt)
                    s[mt][kt] = __builtin_amdgcn_wmma_f32_16x16x32_bf16(
                        false, qa[mt][dk].v, false, kb.v, (short)0, s[mt][kt], false, false);
            }

        // ---- online softmax over this 32-key chunk ----
        float pv[2][2][8];
#pragma unroll
        for (int mt = 0; mt < 2; ++mt)
#pragma unroll
            for (int r = 0; r < 8; ++r) {
                float s0 = s[mt][0][r] * 0.125f;
                float s1 = s[mt][1][r] * 0.125f;
                float v = fmaxf(s0, s1);
                v = fmaxf(v, __shfl_xor(v, 1, 32));
                v = fmaxf(v, __shfl_xor(v, 2, 32));
                v = fmaxf(v, __shfl_xor(v, 4, 32));
                v = fmaxf(v, __shfl_xor(v, 8, 32));        // row max over 32 keys
                float mo   = mrun[mt][r];
                float nm   = fmaxf(mo, v);
                float corr = exp2f((mo - nm) * L2E);
                float p0   = exp2f((s0 - nm) * L2E);
                float p1   = exp2f((s1 - nm) * L2E);
                float rs   = p0 + p1;
                rs += __shfl_xor(rs, 1, 32);
                rs += __shfl_xor(rs, 2, 32);
                rs += __shfl_xor(rs, 4, 32);
                rs += __shfl_xor(rs, 8, 32);               // row sum over 32 keys
                mrun[mt][r] = nm;
                lrun[mt][r] = lrun[mt][r] * corr + rs;
#pragma unroll
                for (int dt = 0; dt < 4; ++dt) o[mt][dt][r] *= corr;
                pv[mt][0][r] = p0;
                pv[mt][1][r] = p1;
            }

        // ---- P: C-layout -> A-layout via LDS (2 KB) ----
        __syncthreads();
#pragma unroll
        for (int mt = 0; mt < 2; ++mt)
#pragma unroll
            for (int kt = 0; kt < 2; ++kt)
#pragma unroll
                for (int r = 0; r < 8; ++r)
                    Pl[(mt * 16 + r + 8 * hi) * 32 + kt * 16 + lid] = (__bf16)pv[mt][kt][r];
        __syncthreads();

        Frag pa[2];
#pragma unroll
        for (int mt = 0; mt < 2; ++mt) {
            pa[mt].h[0] = *(const v8bf*)&Pl[(mt * 16 + lid) * 32 + hi * 8];
            pa[mt].h[1] = *(const v8bf*)&Pl[(mt * 16 + lid) * 32 + 16 + hi * 8];
        }

        // ---- O += P @ V : V is pre-transposed [d][s], contiguous B frags ----
#pragma unroll
        for (int dt = 0; dt < 4; ++dt) {
            Frag vb;
            const __bf16* vp = Vp + (size_t)(dt * 16 + lid) * SEQ + kk0 + hi * 8;
            vb.h[0] = *(const v8bf*)vp;
            vb.h[1] = *(const v8bf*)(vp + 16);
#pragma unroll
            for (int mt = 0; mt < 2; ++mt)
                o[mt][dt] = __builtin_amdgcn_wmma_f32_16x16x32_bf16(
                    false, pa[mt].v, false, vb.v, (short)0, o[mt][dt], false, false);
        }
    }

    // ---- normalize and write att[b][s][h*64+d] ----
#pragma unroll
    for (int mt = 0; mt < 2; ++mt) {
        float linv[8];
#pragma unroll
        for (int r = 0; r < 8; ++r) linv[r] = 1.0f / lrun[mt][r];
#pragma unroll
        for (int dt = 0; dt < 4; ++dt)
#pragma unroll
            for (int r = 0; r < 8; ++r) {
                int srow = q0 + mt * 16 + r + 8 * hi;
                int col  = head * HDIM + dt * 16 + lid;
                att[((size_t)batch * SEQ + srow) * EMBED + col] = (__bf16)(o[mt][dt][r] * linv[r]);
            }
    }
}

// ---------------------------------------------------------------------------
// Host launcher
// ---------------------------------------------------------------------------
extern "C" void kernel_launch(void* const* d_in, const int* in_sizes, int n_in,
                              void* d_out, int out_size, void* d_ws, size_t ws_size,
                              hipStream_t stream)
{
    const float* x  = (const float*)d_in[0];
    const float* Wq = (const float*)d_in[1];  const float* bq = (const float*)d_in[2];
    const float* Aq = (const float*)d_in[3];  const float* Bq = (const float*)d_in[4];
    const float* Wk = (const float*)d_in[5];  const float* bk = (const float*)d_in[6];
    const float* Ak = (const float*)d_in[7];  const float* Bk = (const float*)d_in[8];
    const float* Wv = (const float*)d_in[9];  const float* bv = (const float*)d_in[10];
    const float* Av = (const float*)d_in[11]; const float* Bv = (const float*)d_in[12];
    const float* Wo = (const float*)d_in[13]; const float* bo = (const float*)d_in[14];
    const float* Ao = (const float*)d_in[15]; const float* Bo = (const float*)d_in[16];

    char*  ws  = (char*)d_ws;
    size_t off = 0;
    auto alloc = [&](size_t bytes) -> void* {
        void* p = ws + off;
        off = (off + bytes + 255) & ~(size_t)255;
        return p;
    };
    __bf16* xb  = (__bf16*)alloc((size_t)MTOT * EMBED * 2);      // x in bf16
    __bf16* wq  = (__bf16*)alloc((size_t)EMBED * EMBED * 2);
    __bf16* wk  = (__bf16*)alloc((size_t)EMBED * EMBED * 2);
    __bf16* wv  = (__bf16*)alloc((size_t)EMBED * EMBED * 2);
    __bf16* wo  = (__bf16*)alloc((size_t)EMBED * EMBED * 2);
    __bf16* Qhp = (__bf16*)alloc((size_t)MTOT * EMBED * 2);      // [b][h][s][d]
    __bf16* Khp = (__bf16*)alloc((size_t)MTOT * EMBED * 2);      // [b][h][s][d]
    __bf16* Vtp = (__bf16*)alloc((size_t)MTOT * EMBED * 2);      // [b][h][d][s]
    __bf16* attp= (__bf16*)alloc((size_t)MTOT * EMBED * 2);      // [b][s][e]

    // 1) cast x to bf16
    cvt_bf16_kernel<<<(MTOT * EMBED) / (256 * 4), 256, 0, stream>>>(x, xb, MTOT * EMBED);

    // 2) fold LoRA into effective weights
    make_weff_kernel<<<(EMBED * EMBED) / 256, 256, 0, stream>>>(Wq, Aq, Bq, wq);
    make_weff_kernel<<<(EMBED * EMBED) / 256, 256, 0, stream>>>(Wk, Ak, Bk, wk);
    make_weff_kernel<<<(EMBED * EMBED) / 256, 256, 0, stream>>>(Wv, Av, Bv, wv);
    make_weff_kernel<<<(EMBED * EMBED) / 256, 256, 0, stream>>>(Wo, Ao, Bo, wo);

    // 3) Q/K/V projections (WMMA GEMMs, async-LDS pipelined)
    dim3 gg(EMBED / 128, MTOT / 128);
    gemm_bf16_wmma<<<gg, 128, 0, stream>>>(xb, wq, bq, nullptr, Qhp, 1);
    gemm_bf16_wmma<<<gg, 128, 0, stream>>>(xb, wk, bk, nullptr, Khp, 1);
    gemm_bf16_wmma<<<gg, 128, 0, stream>>>(xb, wv, bv, nullptr, Vtp, 2);

    // 4) flash attention
    flash_attn_kernel<<<dim3(SEQ / 32, 2 * HEADS), 32, 0, stream>>>(Qhp, Khp, Vtp, attp);

    // 5) output projection -> f32 d_out
    gemm_bf16_wmma<<<gg, 128, 0, stream>>>(attp, wo, bo, (float*)d_out, nullptr, 0);
}
